// GnnEncoder_26594437497004
// MI455X (gfx1250) — matrix operand order repaired and checked by reading, not compiled, not run on hardware
//
#include <hip/hip_runtime.h>
#include <hip/hip_bf16.h>

// ---------------- problem constants (match reference) ----------------
#define BB     8
#define NSEQ   4096
#define NN     (BB * NSEQ)        // 32768 nodes
#define DIM    128                // D == DE == 128
#define HD     384                // 2*D + DE
#define FFD    512                // 4*D
#define NEDGE  262144
#define NLAYER 2

typedef __bf16 bf16;
typedef bf16  v8bf  __attribute__((ext_vector_type(8)));
typedef bf16  v16bf __attribute__((ext_vector_type(16)));
typedef float v8f   __attribute__((ext_vector_type(8)));

// ---------------- helpers ----------------
__device__ __forceinline__ bf16 f2bf(float f) {
  union { float f; unsigned u; } a; a.f = f;
  unsigned r = a.u + 0x7FFFu + ((a.u >> 16) & 1u);   // round-nearest-even
  union { unsigned short s; bf16 b; } o; o.s = (unsigned short)(r >> 16);
  return o.b;
}

__device__ __forceinline__ v16bf ld16(const bf16* p) {
  union { v16bf v; v8bf h[2]; } u;
  u.h[0] = *(const v8bf*)(p);
  u.h[1] = *(const v8bf*)(p + 8);
  return u.v;
}

// A-fragment for V_WMMA_F32_16X16X32_BF16 from a row-major bf16 row in LDS.
// lane<16:  elems 0-15 = K[k0..k0+15]
// lane>=16: elems 0-15 = K[k0+16..k0+31]
__device__ __forceinline__ v16bf ld_afrag(const bf16* row, int k0, int hf) {
  union { v16bf v; v8bf h[2]; } u;
  u.h[0] = *(const v8bf*)(row + k0 + hf * 16);
  u.h[1] = *(const v8bf*)(row + k0 + hf * 16 + 8);
  return u.v;
}

__device__ __forceinline__ v8f wmma_bf16(v16bf a, v16bf b, v8f c) {
  return __builtin_amdgcn_wmma_f32_16x16x32_bf16(false, a, false, b, (short)0, c,
                                                 false, false);
}

__device__ __forceinline__ float wsum32(float v) {
  #pragma unroll
  for (int o = 16; o > 0; o >>= 1) v += __shfl_xor(v, o, 32);
  return v;
}

// -------- CDNA5 async DMA: per-lane gather of 16B from global into LDS --------
// each lane's LDS destination offset comes from a VGPR; generic shared-pointer low 32 bits give
// the in-LDS byte offset (flat->LDS mapping truncates to addr[31:0]).
__device__ __forceinline__ unsigned ldsoff(const void* p) {
  return (unsigned)(unsigned long long)p;
}
__device__ __forceinline__ void async_b128(unsigned lds_byte_off, const bf16* g) {
  asm volatile("global_load_async_to_lds_b128 %0, %1, off"
               :: "v"(lds_byte_off), "v"(g) : "memory");
}
__device__ __forceinline__ void wait_async0() {
  asm volatile("s_wait_asynccnt 0" ::: "memory");
}

// ---------------- tiny utility kernels ----------------
__global__ __launch_bounds__(256) void k_zero(float* __restrict__ p, int n) {
  int i = blockIdx.x * 256 + threadIdx.x;
  if (i < n) p[i] = 0.0f;
}

__global__ __launch_bounds__(256) void k_deg(const int* __restrict__ ei,
                                             float* __restrict__ deg) {
  int i = blockIdx.x * 256 + threadIdx.x;
  if (i < NEDGE) atomicAdd(deg + ei[NEDGE + i], 1.0f);
}

__global__ __launch_bounds__(256) void k_invdeg(const float* __restrict__ deg,
                                                float* __restrict__ invdeg) {
  int i = blockIdx.x * 256 + threadIdx.x;
  if (i < NN) invdeg[i] = 1.0f / fmaxf(deg[i], 1.0f);
}

// transpose + convert: W[K][N] f32 -> Wt[N][K] bf16
__global__ __launch_bounds__(256) void k_cvt(const float* __restrict__ W,
                                             bf16* __restrict__ Wt, int K, int N) {
  int i = blockIdx.x * 256 + threadIdx.x;
  if (i < K * N) {
    int k = i / N, n = i - k * N;
    Wt[n * K + k] = f2bf(W[i]);
  }
}

// flat convert f32 -> bf16
__global__ __launch_bounds__(256) void k_cvt_flat(const float* __restrict__ in,
                                                  bf16* __restrict__ out, int n) {
  int i = blockIdx.x * 256 + threadIdx.x;
  if (i < n) out[i] = f2bf(in[i]);
}

// out[row] = LN(x[row] + y[row]) * g + b  (one wave per row); also bf16 mirror.
__global__ __launch_bounds__(256) void k_add_ln(const float* __restrict__ x,
                                                const float* __restrict__ y,
                                                const float* __restrict__ g,
                                                const float* __restrict__ b,
                                                float* __restrict__ out,
                                                bf16* __restrict__ out_bf,
                                                int nrows) {
  int wave = threadIdx.x >> 5, lane = threadIdx.x & 31;
  int row = blockIdx.x * 8 + wave;
  if (row >= nrows) return;
  const float* xr = x + (size_t)row * DIM;
  const float* yr = y + (size_t)row * DIM;
  float v[4]; float s = 0.0f;
  #pragma unroll
  for (int c = 0; c < 4; ++c) { v[c] = xr[lane * 4 + c] + yr[lane * 4 + c]; s += v[c]; }
  s = wsum32(s);
  float mean = s * (1.0f / DIM);
  float q = 0.0f;
  #pragma unroll
  for (int c = 0; c < 4; ++c) { float d = v[c] - mean; q += d * d; }
  q = wsum32(q);
  float inv = rsqrtf(q * (1.0f / DIM) + 1e-5f);
  float* orow = out + (size_t)row * DIM;
  bf16*  brow = out_bf + (size_t)row * DIM;
  #pragma unroll
  for (int c = 0; c < 4; ++c) {
    float o = (v[c] - mean) * inv * g[lane * 4 + c] + b[lane * 4 + c];
    orow[lane * 4 + c] = o;
    brow[lane * 4 + c] = f2bf(o);
  }
}

// async-gather h = [nodes_bf[src] | edge_bf | nodes_bf[dst]] into LDS (64 x 384)
__device__ __forceinline__ void gather_h(bf16* sh_h, const bf16* __restrict__ nodes_bf,
                                         const bf16* __restrict__ edge_bf,
                                         const int* s_src, const int* s_dst,
                                         int e0, int tid) {
  // 64 rows x 48 chunks of 16B (8 bf16) each = 3072 chunks, 12 per thread
  for (int q = tid; q < 64 * 48; q += 256) {
    int r = q / 48;
    int c = q - r * 48;                 // 0..47
    const bf16* g;
    if (c < 16)      g = nodes_bf + (size_t)s_src[r] * DIM + c * 8;
    else if (c < 32) g = edge_bf + (size_t)(e0 + r) * DIM + (c - 16) * 8;
    else             g = nodes_bf + (size_t)s_dst[r] * DIM + (c - 32) * 8;
    async_b128(ldsoff(sh_h + r * HD + c * 8), g);
  }
  wait_async0();
}

// ---------------- message MLP + mean-aggregate scatter ----------------
// 64 edges / block, 8 waves; wave owns 16 rows x 64 cols (4 accumulators).
__global__ __launch_bounds__(256) void k_msg(
    const bf16* __restrict__ nodes_bf, const bf16* __restrict__ edge_bf,
    const int* __restrict__ ei,
    const bf16* __restrict__ w0t, const float* __restrict__ b0,
    const bf16* __restrict__ w1t, const float* __restrict__ b1,
    const bf16* __restrict__ w2t, const float* __restrict__ b2,
    const float* __restrict__ invdeg, float* __restrict__ agg) {
  extern __shared__ char smem[];
  bf16* sh_h  = (bf16*)smem;                                // 64 x 384
  bf16* sh_m0 = (bf16*)(smem + 64 * HD * 2);                // 64 x 128
  bf16* sh_m1 = (bf16*)(smem + 64 * HD * 2 + 64 * DIM * 2); // 64 x 128
  __shared__ int s_src[64], s_dst[64];

  const int tid = threadIdx.x;
  const int e0  = blockIdx.x * 64;
  if (tid < 64) { s_src[tid] = ei[e0 + tid]; s_dst[tid] = ei[NEDGE + e0 + tid]; }
  __syncthreads();

  gather_h(sh_h, nodes_bf, edge_bf, s_src, s_dst, e0, tid);
  __syncthreads();

  const int wave = tid >> 5, lane = tid & 31;
  const int hf = lane >> 4, ln = lane & 15;
  const int rtile = wave & 3;
  const int rb = rtile * 16 + hf * 8;
  const int cb = (wave >> 2) * 64;     // base col of this wave's 4 tiles

  // ---- layer 0: K=384, relu (k-outer, 4 live accumulators) ----
  {
    const bf16* arow = sh_h + (rtile * 16 + ln) * HD;
    v8f acc[4];
    #pragma unroll
    for (int j = 0; j < 4; ++j) acc[j] = 0.0f;
    #pragma unroll
    for (int k0 = 0; k0 < HD; k0 += 32) {
      v16bf a = ld_afrag(arow, k0, hf);
      #pragma unroll
      for (int j = 0; j < 4; ++j)
        acc[j] = wmma_bf16(a, ld16(w0t + (cb + j * 16 + ln) * HD + k0 + hf * 16),
                           acc[j]);
    }
    #pragma unroll
    for (int j = 0; j < 4; ++j) {
      const int col = cb + j * 16 + ln;
      const float bias = b0[col];
      #pragma unroll
      for (int r = 0; r < 8; ++r) {
        float v = acc[j][r] + bias;
        sh_m0[(rb + r) * DIM + col] = f2bf(v > 0.0f ? v : 0.0f);
      }
    }
  }
  __syncthreads();

  // ---- layer 1: K=128, relu (A fragments hoisted) ----
  {
    const bf16* arow = sh_m0 + (rtile * 16 + ln) * DIM;
    v16bf afr[4];
    #pragma unroll
    for (int k = 0; k < 4; ++k) afr[k] = ld_afrag(arow, k * 32, hf);
    v8f acc[4];
    #pragma unroll
    for (int j = 0; j < 4; ++j) acc[j] = 0.0f;
    #pragma unroll
    for (int k = 0; k < 4; ++k)
      #pragma unroll
      for (int j = 0; j < 4; ++j)
        acc[j] = wmma_bf16(afr[k],
                           ld16(w1t + (cb + j * 16 + ln) * DIM + k * 32 + hf * 16),
                           acc[j]);
    #pragma unroll
    for (int j = 0; j < 4; ++j) {
      const int col = cb + j * 16 + ln;
      const float bias = b1[col];
      #pragma unroll
      for (int r = 0; r < 8; ++r) {
        float v = acc[j][r] + bias;
        sh_m1[(rb + r) * DIM + col] = f2bf(v > 0.0f ? v : 0.0f);
      }
    }
  }
  __syncthreads();

  // ---- layer 2: K=128, scaled atomic scatter to agg ----
  {
    const bf16* arow = sh_m1 + (rtile * 16 + ln) * DIM;
    v16bf afr[4];
    #pragma unroll
    for (int k = 0; k < 4; ++k) afr[k] = ld_afrag(arow, k * 32, hf);
    v8f acc[4];
    #pragma unroll
    for (int j = 0; j < 4; ++j) acc[j] = 0.0f;
    #pragma unroll
    for (int k = 0; k < 4; ++k)
      #pragma unroll
      for (int j = 0; j < 4; ++j)
        acc[j] = wmma_bf16(afr[k],
                           ld16(w2t + (cb + j * 16 + ln) * DIM + k * 32 + hf * 16),
                           acc[j]);
    #pragma unroll
    for (int j = 0; j < 4; ++j) {
      const int col = cb + j * 16 + ln;
      const float bias = b2[col];
      #pragma unroll
      for (int r = 0; r < 8; ++r) {
        const int d = s_dst[rb + r];
        atomicAdd(agg + (size_t)d * DIM + col, (acc[j][r] + bias) * invdeg[d]);
      }
    }
  }
}

// ---------------- feedforward: 128 -> 512 (relu) -> 128 ----------------
__global__ __launch_bounds__(256) void k_ff(
    const bf16* __restrict__ nodes_bf,
    const bf16* __restrict__ w0t, const float* __restrict__ b0,
    const bf16* __restrict__ w1t, const float* __restrict__ b1,
    float* __restrict__ out) {
  extern __shared__ char smem[];
  bf16* sh_x   = (bf16*)smem;                    // 64 x 128
  bf16* sh_hid = (bf16*)(smem + 64 * DIM * 2);   // 64 x 512

  const int tid = threadIdx.x;
  const int n0  = blockIdx.x * 64;
  // contiguous async copy: 1024 chunks of 16B, 4 per thread
  for (int q = tid; q < 1024; q += 256)
    async_b128(ldsoff(sh_x + q * 8), nodes_bf + (size_t)n0 * DIM + q * 8);
  wait_async0();
  __syncthreads();

  const int wave = tid >> 5, lane = tid & 31;
  const int hf = lane >> 4, ln = lane & 15;
  const int rtile = wave & 3;
  const int rb = rtile * 16 + hf * 8;

  // ---- layer 0: [64x128] x [128x512], relu; A hoisted, 4 groups of 4 tiles ----
  {
    const bf16* arow = sh_x + (rtile * 16 + ln) * DIM;
    v16bf afr[4];
    #pragma unroll
    for (int k = 0; k < 4; ++k) afr[k] = ld_afrag(arow, k * 32, hf);
    #pragma unroll
    for (int grp = 0; grp < 4; ++grp) {
      const int cb = (wave >> 2) * 256 + grp * 64;
      v8f acc[4];
      #pragma unroll
      for (int j = 0; j < 4; ++j) acc[j] = 0.0f;
      #pragma unroll
      for (int k = 0; k < 4; ++k)
        #pragma unroll
        for (int j = 0; j < 4; ++j)
          acc[j] = wmma_bf16(afr[k],
                             ld16(w0t + (cb + j * 16 + ln) * DIM + k * 32 + hf * 16),
                             acc[j]);
      #pragma unroll
      for (int j = 0; j < 4; ++j) {
        const int col = cb + j * 16 + ln;
        const float bias = b0[col];
        #pragma unroll
        for (int r = 0; r < 8; ++r) {
          float v = acc[j][r] + bias;
          sh_hid[(rb + r) * FFD + col] = f2bf(v > 0.0f ? v : 0.0f);
        }
      }
    }
  }
  __syncthreads();

  // ---- layer 1: [64x512] x [512x128] (k-outer, 4 accumulators) ----
  {
    const bf16* arow = sh_hid + (rtile * 16 + ln) * FFD;
    const int cb = (wave >> 2) * 64;
    v8f acc[4];
    #pragma unroll
    for (int j = 0; j < 4; ++j) acc[j] = 0.0f;
    #pragma unroll
    for (int k0 = 0; k0 < FFD; k0 += 32) {
      v16bf a = ld_afrag(arow, k0, hf);
      #pragma unroll
      for (int j = 0; j < 4; ++j)
        acc[j] = wmma_bf16(a, ld16(w1t + (cb + j * 16 + ln) * FFD + k0 + hf * 16),
                           acc[j]);
    }
    #pragma unroll
    for (int j = 0; j < 4; ++j) {
      const int col = cb + j * 16 + ln;
      const float bias = b1[col];
      #pragma unroll
      for (int r = 0; r < 8; ++r)
        out[(size_t)(n0 + rb + r) * DIM + col] = acc[j][r] + bias;
    }
  }
}

// ---------------- edge-update MLP + in-block residual LayerNorm ----------------
__global__ __launch_bounds__(256) void k_edge(
    const bf16* __restrict__ nodes_bf, float* __restrict__ edge,
    bf16* __restrict__ edge_bf, const int* __restrict__ ei,
    const bf16* __restrict__ w0t, const float* __restrict__ b0,
    const bf16* __restrict__ w1t, const float* __restrict__ b1,
    const bf16* __restrict__ w2t, const float* __restrict__ b2,
    const float* __restrict__ g, const float* __restrict__ bln) {
  extern __shared__ char smem[];
  bf16*  sh_h  = (bf16*)smem;                                // 64 x 384
  bf16*  sh_m0 = (bf16*)(smem + 64 * HD * 2);                // 64 x 128
  bf16*  sh_m1 = (bf16*)(smem + 64 * HD * 2 + 64 * DIM * 2); // 64 x 128
  float* sh_e  = (float*)smem;  // overlay on dead h region (32KB < 48KB)
  __shared__ int s_src[64], s_dst[64];

  const int tid = threadIdx.x;
  const int e0  = blockIdx.x * 64;
  if (tid < 64) { s_src[tid] = ei[e0 + tid]; s_dst[tid] = ei[NEDGE + e0 + tid]; }
  __syncthreads();

  gather_h(sh_h, nodes_bf, edge_bf, s_src, s_dst, e0, tid);
  __syncthreads();

  const int wave = tid >> 5, lane = tid & 31;
  const int hf = lane >> 4, ln = lane & 15;
  const int rtile = wave & 3;
  const int rb = rtile * 16 + hf * 8;
  const int cb = (wave >> 2) * 64;

  // ---- layer 0 (K=384, relu) ----
  {
    const bf16* arow = sh_h + (rtile * 16 + ln) * HD;
    v8f acc[4];
    #pragma unroll
    for (int j = 0; j < 4; ++j) acc[j] = 0.0f;
    #pragma unroll
    for (int k0 = 0; k0 < HD; k0 += 32) {
      v16bf a = ld_afrag(arow, k0, hf);
      #pragma unroll
      for (int j = 0; j < 4; ++j)
        acc[j] = wmma_bf16(a, ld16(w0t + (cb + j * 16 + ln) * HD + k0 + hf * 16),
                           acc[j]);
    }
    #pragma unroll
    for (int j = 0; j < 4; ++j) {
      const int col = cb + j * 16 + ln;
      const float bias = b0[col];
      #pragma unroll
      for (int r = 0; r < 8; ++r) {
        float v = acc[j][r] + bias;
        sh_m0[(rb + r) * DIM + col] = f2bf(v > 0.0f ? v : 0.0f);
      }
    }
  }
  __syncthreads();

  // ---- layer 1 (K=128, relu) ----
  {
    const bf16* arow = sh_m0 + (rtile * 16 + ln) * DIM;
    v16bf afr[4];
    #pragma unroll
    for (int k = 0; k < 4; ++k) afr[k] = ld_afrag(arow, k * 32, hf);
    v8f acc[4];
    #pragma unroll
    for (int j = 0; j < 4; ++j) acc[j] = 0.0f;
    #pragma unroll
    for (int k = 0; k < 4; ++k)
      #pragma unroll
      for (int j = 0; j < 4; ++j)
        acc[j] = wmma_bf16(afr[k],
                           ld16(w1t + (cb + j * 16 + ln) * DIM + k * 32 + hf * 16),
                           acc[j]);
    #pragma unroll
    for (int j = 0; j < 4; ++j) {
      const int col = cb + j * 16 + ln;
      const float bias = b1[col];
      #pragma unroll
      for (int r = 0; r < 8; ++r) {
        float v = acc[j][r] + bias;
        sh_m1[(rb + r) * DIM + col] = f2bf(v > 0.0f ? v : 0.0f);
      }
    }
  }
  __syncthreads();

  // ---- layer 2 (K=128) -> f32 e-tile in LDS ----
  {
    const bf16* arow = sh_m1 + (rtile * 16 + ln) * DIM;
    v16bf afr[4];
    #pragma unroll
    for (int k = 0; k < 4; ++k) afr[k] = ld_afrag(arow, k * 32, hf);
    v8f acc[4];
    #pragma unroll
    for (int j = 0; j < 4; ++j) acc[j] = 0.0f;
    #pragma unroll
    for (int k = 0; k < 4; ++k)
      #pragma unroll
      for (int j = 0; j < 4; ++j)
        acc[j] = wmma_bf16(afr[k],
                           ld16(w2t + (cb + j * 16 + ln) * DIM + k * 32 + hf * 16),
                           acc[j]);
    #pragma unroll
    for (int j = 0; j < 4; ++j) {
      const int col = cb + j * 16 + ln;
      const float bias = b2[col];
      #pragma unroll
      for (int r = 0; r < 8; ++r)
        sh_e[(rb + r) * DIM + col] = acc[j][r] + bias;
    }
  }
  __syncthreads();

  // ---- residual + LayerNorm, one wave per edge row (8 rows/wave) ----
  for (int r8 = 0; r8 < 8; ++r8) {
    const int row = wave * 8 + r8;
    const float* er = edge + (size_t)(e0 + row) * DIM;
    float v[4]; float s = 0.0f;
    #pragma unroll
    for (int c = 0; c < 4; ++c) {
      v[c] = er[lane * 4 + c] + sh_e[row * DIM + lane * 4 + c];
      s += v[c];
    }
    s = wsum32(s);
    const float mean = s * (1.0f / DIM);
    float q = 0.0f;
    #pragma unroll
    for (int c = 0; c < 4; ++c) { float d = v[c] - mean; q += d * d; }
    q = wsum32(q);
    const float inv = rsqrtf(q * (1.0f / DIM) + 1e-5f);
    float* orow = edge + (size_t)(e0 + row) * DIM;
    bf16*  brow = edge_bf + (size_t)(e0 + row) * DIM;
    #pragma unroll
    for (int c = 0; c < 4; ++c) {
      float o = (v[c] - mean) * inv * g[lane * 4 + c] + bln[lane * 4 + c];
      orow[lane * 4 + c] = o;
      brow[lane * 4 + c] = f2bf(o);
    }
  }
}

// ---------------- host orchestration ----------------
extern "C" void kernel_launch(void* const* d_in, const int* in_sizes, int n_in,
                              void* d_out, int out_size, void* d_ws, size_t ws_size,
                              hipStream_t stream) {
  const float* x        = (const float*)d_in[0];
  const int*   ei       = (const int*)d_in[1];
  const float* ea       = (const float*)d_in[2];
  const float* msg_W0   = (const float*)d_in[3];
  const float* msg_b0   = (const float*)d_in[4];
  const float* msg_W1   = (const float*)d_in[5];
  const float* msg_b1   = (const float*)d_in[6];
  const float* msg_W2   = (const float*)d_in[7];
  const float* msg_b2   = (const float*)d_in[8];
  const float* norm0_g  = (const float*)d_in[9];
  const float* norm0_b  = (const float*)d_in[10];
  const float* ff_W0    = (const float*)d_in[11];
  const float* ff_b0    = (const float*)d_in[12];
  const float* ff_W1    = (const float*)d_in[13];
  const float* ff_b1    = (const float*)d_in[14];
  const float* norm1_g  = (const float*)d_in[15];
  const float* norm1_b  = (const float*)d_in[16];
  const float* edge_W0  = (const float*)d_in[17];
  const float* edge_b0  = (const float*)d_in[18];
  const float* edge_W1  = (const float*)d_in[19];
  const float* edge_b1  = (const float*)d_in[20];
  const float* edge_W2  = (const float*)d_in[21];
  const float* edge_b2  = (const float*)d_in[22];
  const float* enorm_g  = (const float*)d_in[23];
  const float* enorm_b  = (const float*)d_in[24];

  // workspace layout
  float* nodes  = (float*)d_ws;                    // NN*DIM f32
  float* agg    = nodes + (size_t)NN * DIM;        // NN*DIM f32 (also FF out)
  float* edge   = agg + (size_t)NN * DIM;          // NEDGE*DIM f32
  float* invdeg = edge + (size_t)NEDGE * DIM;      // NN
  float* deg    = invdeg + NN;                     // NN
  bf16* nodes_bf = (bf16*)(deg + NN);              // NN*DIM bf16
  bf16* edge_bf  = nodes_bf + (size_t)NN * DIM;    // NEDGE*DIM bf16
  bf16* wbase    = edge_bf + (size_t)NEDGE * DIM;

  // per-layer bf16 transposed weights
  const size_t SZ_M0 = (size_t)DIM * HD;
  const size_t SZ_SQ = (size_t)DIM * DIM;
  const size_t SZ_F0 = (size_t)FFD * DIM;
  const size_t SZ_F1 = (size_t)DIM * FFD;
  const size_t PER_L = SZ_M0 + 2 * SZ_SQ + SZ_F0 + SZ_F1 + SZ_M0 + 2 * SZ_SQ;

  bf16* m0t[NLAYER]; bf16* m1t[NLAYER]; bf16* m2t[NLAYER];
  bf16* f0t[NLAYER]; bf16* f1t[NLAYER];
  bf16* e0t[NLAYER]; bf16* e1t[NLAYER]; bf16* e2t[NLAYER];
  for (int l = 0; l < NLAYER; ++l) {
    bf16* p = wbase + (size_t)l * PER_L;
    m0t[l] = p; p += SZ_M0;
    m1t[l] = p; p += SZ_SQ;
    m2t[l] = p; p += SZ_SQ;
    f0t[l] = p; p += SZ_F0;
    f1t[l] = p; p += SZ_F1;
    e0t[l] = p; p += SZ_M0;
    e1t[l] = p; p += SZ_SQ;
    e2t[l] = p; p += SZ_SQ;
  }

  const int SMEM_MSG  = (64 * HD + 2 * 64 * DIM) * 2;  // 81920 B
  const int SMEM_FF   = (64 * DIM + 64 * FFD) * 2;     // 81920 B
  const int SMEM_EDGE = SMEM_MSG;
  (void)hipFuncSetAttribute((const void*)k_msg,
      hipFuncAttributeMaxDynamicSharedMemorySize, SMEM_MSG);
  (void)hipFuncSetAttribute((const void*)k_ff,
      hipFuncAttributeMaxDynamicSharedMemorySize, SMEM_FF);
  (void)hipFuncSetAttribute((const void*)k_edge,
      hipFuncAttributeMaxDynamicSharedMemorySize, SMEM_EDGE);

  // weight conversion (transpose to [N][K] bf16)
  for (int l = 0; l < NLAYER; ++l) {
    k_cvt<<<(HD * DIM + 255) / 256, 256, 0, stream>>>(msg_W0 + (size_t)l * HD * DIM, m0t[l], HD, DIM);
    k_cvt<<<(DIM * DIM + 255) / 256, 256, 0, stream>>>(msg_W1 + (size_t)l * DIM * DIM, m1t[l], DIM, DIM);
    k_cvt<<<(DIM * DIM + 255) / 256, 256, 0, stream>>>(msg_W2 + (size_t)l * DIM * DIM, m2t[l], DIM, DIM);
    k_cvt<<<(DIM * FFD + 255) / 256, 256, 0, stream>>>(ff_W0 + (size_t)l * DIM * FFD, f0t[l], DIM, FFD);
    k_cvt<<<(FFD * DIM + 255) / 256, 256, 0, stream>>>(ff_W1 + (size_t)l * FFD * DIM, f1t[l], FFD, DIM);
    k_cvt<<<(HD * DIM + 255) / 256, 256, 0, stream>>>(edge_W0 + (size_t)l * HD * DIM, e0t[l], HD, DIM);
    k_cvt<<<(DIM * DIM + 255) / 256, 256, 0, stream>>>(edge_W1 + (size_t)l * DIM * DIM, e1t[l], DIM, DIM);
    k_cvt<<<(DIM * DIM + 255) / 256, 256, 0, stream>>>(edge_W2 + (size_t)l * DIM * DIM, e2t[l], DIM, DIM);
  }

  // working copies + bf16 mirrors
  hipMemcpyAsync(nodes, x, (size_t)NN * DIM * sizeof(float),
                 hipMemcpyDeviceToDevice, stream);
  hipMemcpyAsync(edge, ea, (size_t)NEDGE * DIM * sizeof(float),
                 hipMemcpyDeviceToDevice, stream);
  k_cvt_flat<<<(NN * DIM) / 256, 256, 0, stream>>>(x, nodes_bf, NN * DIM);
  k_cvt_flat<<<(NEDGE * DIM) / 256, 256, 0, stream>>>(ea, edge_bf, NEDGE * DIM);

  // degrees
  k_zero<<<(NN + 255) / 256, 256, 0, stream>>>(deg, NN);
  k_deg<<<NEDGE / 256, 256, 0, stream>>>(ei, deg);
  k_invdeg<<<NN / 256, 256, 0, stream>>>(deg, invdeg);

  for (int l = 0; l < NLAYER; ++l) {
    // message passing
    k_zero<<<(NN * DIM) / 256, 256, 0, stream>>>(agg, NN * DIM);
    k_msg<<<NEDGE / 64, 256, SMEM_MSG, stream>>>(
        nodes_bf, edge_bf, ei,
        m0t[l], msg_b0 + (size_t)l * DIM,
        m1t[l], msg_b1 + (size_t)l * DIM,
        m2t[l], msg_b2 + (size_t)l * DIM,
        invdeg, agg);
    k_add_ln<<<NN / 8, 256, 0, stream>>>(nodes, agg,
        norm0_g + (size_t)l * DIM, norm0_b + (size_t)l * DIM, nodes, nodes_bf, NN);

    // feedforward (agg reused as FF output buffer)
    k_ff<<<NN / 64, 256, SMEM_FF, stream>>>(
        nodes_bf, f0t[l], ff_b0 + (size_t)l * FFD, f1t[l], ff_b1 + (size_t)l * DIM,
        agg);
    k_add_ln<<<NN / 8, 256, 0, stream>>>(nodes, agg,
        norm1_g + (size_t)l * DIM, norm1_b + (size_t)l * DIM, nodes, nodes_bf, NN);

    // edge update (in-place on working edge buffers)
    k_edge<<<NEDGE / 64, 256, SMEM_EDGE, stream>>>(
        nodes_bf, edge, edge_bf, ei,
        e0t[l], edge_b0 + (size_t)l * DIM,
        e1t[l], edge_b1 + (size_t)l * DIM,
        e2t[l], edge_b2 + (size_t)l * DIM,
        enorm_g + (size_t)l * DIM, enorm_b + (size_t)l * DIM);
  }

  hipMemcpyAsync(d_out, nodes, (size_t)NN * DIM * sizeof(float),
                 hipMemcpyDeviceToDevice, stream);
}
